// MultiHeadAttention_34428457844813
// MI455X (gfx1250) — matrix operand order repaired and checked by reading, not compiled, not run on hardware
//
#include <hip/hip_runtime.h>
#include <hip/hip_bf16.h>
#include <stdint.h>

// ---------------------------------------------------------------------------
// MHA forward for MI455X (gfx1250, wave32, WMMA bf16 16x16x32, f32 accum)
//   D=768, H=12, Hd=64, B=2, S=4096 causal
// Attention: 4-wave workgroups; K/V double-buffered in LDS via async-to-LDS
// copies (ASYNCcnt) pipelined against WMMA compute.
// ---------------------------------------------------------------------------

#define D_MODEL  768
#define N_HEADS  12
#define HEAD_DIM 64
#define BATCH    2
#define SEQ      4096
#define BS       (BATCH * SEQ)      // 8192 rows
#define NEGV     (-1.0e9f)

typedef __attribute__((ext_vector_type(16))) __bf16 v16bf;
typedef __attribute__((ext_vector_type(8)))  float  v8f;
typedef __attribute__((ext_vector_type(4)))  int    v4i;

#define AS1 __attribute__((address_space(1)))
#define AS3 __attribute__((address_space(3)))

static __device__ __forceinline__ __bf16 f2bf(float f) { return (__bf16)f; }

static __device__ __forceinline__ v8f wmma_bf16(v16bf a, v16bf b, v8f c) {
    return __builtin_amdgcn_wmma_f32_16x16x32_bf16(false, a, false, b,
                                                   (short)0, c, false, false);
}

// A-fragment k offset for pair p (p = element/2), half hf:
//  VGPR0..3 : K = hf*8 + {0,2,4,6}
//  VGPR4..7 : K = 16 + hf*8 + {0,2,4,6}
static __device__ __forceinline__ int a_koff(int p, int hf) {
    return (p < 4) ? (hf * 8 + 2 * p) : (16 + hf * 8 + 2 * (p - 4));
}

// 16-byte async global->LDS copy (ASYNCcnt-tracked); sync fallback otherwise.
static __device__ __forceinline__ void async_b128(const void* g, void* l) {
#if __has_builtin(__builtin_amdgcn_global_load_async_to_lds_b128)
    __builtin_amdgcn_global_load_async_to_lds_b128(
        (AS1 v4i*)g, (AS3 v4i*)l, 0, 0);
#else
    *(uint4*)l = *(const uint4*)g;
#endif
}

static __device__ __forceinline__ void wait_async() {
#if __has_builtin(__builtin_amdgcn_s_wait_asynccnt)
    __builtin_amdgcn_s_wait_asynccnt(0);
#elif __has_builtin(__builtin_amdgcn_global_load_async_to_lds_b128)
    asm volatile("s_wait_asynccnt 0x0" ::: "memory");
#endif
}

// ---------------------------------------------------------------------------
// Kernel 1: fused Q/K/V projection.  C[m,n] = X[m,:] . W[n,:] + bias[n]
// grid = (BS/16, D/64, 3); block = 32 (one wave); z: 0=Q, 1=K, 2=V
// Q,K stored bf16 [B,H,S,Hd]; V stored transposed bf16 [B,H,Hd,S].
// ---------------------------------------------------------------------------
__global__ void __launch_bounds__(32)
proj_qkv_kernel(const float* __restrict__ Xq, const float* __restrict__ Xk,
                const float* __restrict__ Xv,
                const float* __restrict__ Wq, const float* __restrict__ Wk,
                const float* __restrict__ Wv,
                const float* __restrict__ bq, const float* __restrict__ bk,
                const float* __restrict__ bv,
                __bf16* __restrict__ qb, __bf16* __restrict__ kb,
                __bf16* __restrict__ vt)
{
    const int lane = threadIdx.x & 31;
    const int ln = lane & 15, hf = lane >> 4;
    const int m0 = blockIdx.x * 16;
    const int n0 = blockIdx.y * 64;
    const int z  = blockIdx.z;

    const float* X    = (z == 0) ? Xq : (z == 1) ? Xk : Xv;
    const float* W    = (z == 0) ? Wq : (z == 1) ? Wk : Wv;
    const float* bias = (z == 0) ? bq : (z == 1) ? bk : bv;

    v8f acc[4] = {};
    const int mA = m0 + ln;                       // A-matrix row for this lane

    for (int k0 = 0; k0 < D_MODEL; k0 += 32) {
        v16bf a;
        #pragma unroll
        for (int p = 0; p < 8; ++p) {
            const int kk = a_koff(p, hf);
            const float* src = X + (size_t)mA * D_MODEL + k0 + kk;
            a[2 * p]     = f2bf(src[0]);
            a[2 * p + 1] = f2bf(src[1]);
        }
        #pragma unroll
        for (int nt = 0; nt < 4; ++nt) {
            const int n = n0 + nt * 16 + ln;      // B-matrix column
            v16bf bm;
            #pragma unroll
            for (int p = 0; p < 8; ++p) {
                const int kk = a_koff(p, hf);     // B: same K order as A
                const float* src = W + (size_t)n * D_MODEL + k0 + kk; // W[n,k]
                bm[2 * p]     = f2bf(src[0]);
                bm[2 * p + 1] = f2bf(src[1]);
            }
            acc[nt] = wmma_bf16(a, bm, acc[nt]);
        }
    }

    // Bias add; scatter to head-split layout.
    #pragma unroll
    for (int nt = 0; nt < 4; ++nt) {
        const int n    = n0 + nt * 16 + ln;
        const float bv_ = bias[n];
        const int head = n / HEAD_DIM;
        const int d    = n % HEAD_DIM;
        #pragma unroll
        for (int r = 0; r < 8; ++r) {
            const int row = m0 + r + 8 * hf;      // global row in [0, BS)
            const int bb  = row / SEQ;
            const int s   = row % SEQ;
            const __bf16 o = f2bf(acc[nt][r] + bv_);
            if (z == 0)
                qb[(((size_t)bb * N_HEADS + head) * SEQ + s) * HEAD_DIM + d] = o;
            else if (z == 1)
                kb[(((size_t)bb * N_HEADS + head) * SEQ + s) * HEAD_DIM + d] = o;
            else
                vt[(((size_t)bb * N_HEADS + head) * HEAD_DIM + d) * SEQ + s] = o;
        }
    }
}

// ---------------------------------------------------------------------------
// Kernel 2: causal flash attention.
// grid = (S/64, H, B); block = 128 (4 waves). Each wave owns 16 query rows;
// the workgroup cooperatively double-buffers 32-key K/V blocks in LDS with
// async-to-LDS copies, pipelined one block ahead of the WMMA compute.
// ---------------------------------------------------------------------------
__global__ void __launch_bounds__(128)
attn_kernel(const __bf16* __restrict__ qb, const __bf16* __restrict__ kb,
            const __bf16* __restrict__ vt, __bf16* __restrict__ ctx)
{
    __shared__ __bf16 klds[2][32][HEAD_DIM]; // K blocks: 32 keys x 64 d (2x4KB)
    __shared__ __bf16 vlds[2][HEAD_DIM][32]; // V blocks: 64 d x 32 keys (2x4KB)
    __shared__ __bf16 pbuf[4][16][32];       // per-wave P tile staging (4KB)

    const int tid  = threadIdx.x;           // 0..127
    const int wv   = tid >> 5;              // wave id 0..3
    const int lane = tid & 31;
    const int ln = lane & 15, hf = lane >> 4;
    const int m0 = blockIdx.x * 64;         // workgroup q-row base
    const int mw = m0 + wv * 16;            // this wave's q-row base
    const int h  = blockIdx.y;
    const int b  = blockIdx.z;

    const __bf16* qh = qb + (((size_t)b * N_HEADS + h) * SEQ) * HEAD_DIM;
    const __bf16* kh = kb + (((size_t)b * N_HEADS + h) * SEQ) * HEAD_DIM;
    const __bf16* vh = vt + (((size_t)b * N_HEADS + h) * HEAD_DIM) * SEQ;

    // Issue async copies for the K/V block starting at key j0 into buffer buf.
    auto stage = [&](int j0, int buf) {
        const char* gk = (const char*)(kh + (size_t)j0 * HEAD_DIM); // 4KB contig
        char* lk = (char*)&klds[buf][0][0];
        #pragma unroll
        for (int i = 0; i < 2; ++i) {
            const int off = (tid * 2 + i) * 16;           // 0..4080
            async_b128(gk + off, lk + off);
        }
        #pragma unroll
        for (int i = 0; i < 2; ++i) {
            const int idx = tid * 2 + i;                  // 0..255 chunks
            const int d = idx >> 2, o = (idx & 3) * 16;   // 64B per V row
            const char* gv = (const char*)(vh + (size_t)d * SEQ + j0);
            async_b128(gv + o, (char*)&vlds[buf][d][0] + o);
        }
    };

    // Q A-fragments: two 16x32 chunks over head_dim=64 (pairs contiguous in d)
    v16bf qa[2];
    #pragma unroll
    for (int c = 0; c < 2; ++c) {
        #pragma unroll
        for (int p = 0; p < 8; ++p) {
            const int kk = a_koff(p, hf);
            const __bf16* src = qh + (size_t)(mw + ln) * HEAD_DIM + c * 32 + kk;
            qa[c][2 * p]     = src[0];
            qa[c][2 * p + 1] = src[1];
        }
    }

    v8f acc[4] = {};
    float mrow[8], lrow[8];
    #pragma unroll
    for (int r = 0; r < 8; ++r) { mrow[r] = -3.0e38f; lrow[r] = 0.0f; }

    const int jend = m0 + 64;               // causal bound for this workgroup
    stage(0, 0);                            // prologue: prefetch block 0

    for (int j0 = 0; j0 < jend; j0 += 32) { // causal key blocks of 32
        const int buf = (j0 >> 5) & 1;
        wait_async();                       // own copies for `buf` complete
        __syncthreads();                    // all waves' copies visible; buf^1
                                            // no longer being read by anyone
        if (j0 + 32 < jend)
            stage(j0 + 32, buf ^ 1);        // prefetch next block (overlaps)

        if (j0 <= mw + 15) {                // wave-uniform causal skip
            // ---- scores S = Q . K^T : two 16x16 tiles, 2 WMMAs each ----
            v8f s[2] = {};
            #pragma unroll
            for (int t = 0; t < 2; ++t) {
                #pragma unroll
                for (int c = 0; c < 2; ++c) {
                    v16bf kf;
                    const uint32_t* kro =
                        (const uint32_t*)&klds[buf][t * 16 + ln][0];
                    #pragma unroll
                    for (int p = 0; p < 8; ++p) {
                        const int dd = c * 32 + a_koff(p, hf);
                        union { uint32_t u; __bf16 hv[2]; } cv;
                        cv.u = kro[dd >> 1];
                        kf[2 * p]     = cv.hv[0];
                        kf[2 * p + 1] = cv.hv[1];
                    }
                    s[t] = wmma_bf16(qa[c], kf, s[t]);
                }
            }

            // ---- scale + causal mask + online softmax ----
            #pragma unroll
            for (int r = 0; r < 8; ++r) {
                const int row = mw + r + 8 * hf;
                const int c0 = j0 + ln, c1 = c0 + 16;
                float s0 = s[0][r] * 0.125f;   // 1/sqrt(64)
                float s1 = s[1][r] * 0.125f;
                if (c0 > row) s0 = NEGV;
                if (c1 > row) s1 = NEGV;

                float mx = fmaxf(s0, s1);
                #pragma unroll
                for (int off = 1; off < 16; off <<= 1)
                    mx = fmaxf(mx, __shfl_xor(mx, off, 32));
                const float mnew  = fmaxf(mrow[r], mx);
                const float scale = __expf(mrow[r] - mnew);
                const float p0 = __expf(s0 - mnew);
                const float p1 = __expf(s1 - mnew);
                float rs = p0 + p1;
                #pragma unroll
                for (int off = 1; off < 16; off <<= 1)
                    rs += __shfl_xor(rs, off, 32);
                lrow[r] = lrow[r] * scale + rs;
                mrow[r] = mnew;
                #pragma unroll
                for (int nt = 0; nt < 4; ++nt) acc[nt][r] *= scale;

                pbuf[wv][r + 8 * hf][ln]      = f2bf(p0);
                pbuf[wv][r + 8 * hf][ln + 16] = f2bf(p1);
            }
            asm volatile("s_wait_dscnt 0" ::: "memory");

            // ---- P A-fragment from wave-private LDS tile ----
            v16bf pa;
            const uint32_t* prow = (const uint32_t*)&pbuf[wv][ln][0];
            #pragma unroll
            for (int p = 0; p < 8; ++p) {
                const int kk = a_koff(p, hf);
                union { uint32_t u; __bf16 hv[2]; } cv;
                cv.u = prow[kk >> 1];
                pa[2 * p]     = cv.hv[0];
                pa[2 * p + 1] = cv.hv[1];
            }

            // ---- O += P . V : 4 WMMAs (V rows in LDS, contiguous k pairs) --
            #pragma unroll
            for (int nt = 0; nt < 4; ++nt) {
                v16bf vf;
                const uint32_t* vro =
                    (const uint32_t*)&vlds[buf][nt * 16 + ln][0];
                #pragma unroll
                for (int p = 0; p < 8; ++p) {
                    const int kk = a_koff(p, hf);
                    union { uint32_t u; __bf16 hv[2]; } cv;
                    cv.u = vro[kk >> 1];
                    vf[2 * p]     = cv.hv[0];
                    vf[2 * p + 1] = cv.hv[1];
                }
                acc[nt] = wmma_bf16(pa, vf, acc[nt]);
            }
        }
    }

    // ---- normalize and store ctx bf16 [B, S, D] ----
    #pragma unroll
    for (int r = 0; r < 8; ++r) {
        const float inv = 1.0f / lrow[r];
        const int row = mw + r + 8 * hf;
        #pragma unroll
        for (int nt = 0; nt < 4; ++nt) {
            const int dcol = h * HEAD_DIM + nt * 16 + ln;
            ctx[((size_t)b * SEQ + row) * D_MODEL + dcol] =
                f2bf(acc[nt][r] * inv);
        }
    }
}

// ---------------------------------------------------------------------------
// Kernel 3: output projection. out[m,n] = ctx[m,:] . Wo[n,:] + bo[n] (f32 out)
// grid = (BS/16, D/64); block = 32.
// ---------------------------------------------------------------------------
__global__ void __launch_bounds__(32)
proj_out_kernel(const __bf16* __restrict__ ctx, const float* __restrict__ Wo,
                const float* __restrict__ bo, float* __restrict__ out)
{
    const int lane = threadIdx.x & 31;
    const int ln = lane & 15, hf = lane >> 4;
    const int m0 = blockIdx.x * 16;
    const int n0 = blockIdx.y * 64;

    v8f acc[4] = {};
    const int mA = m0 + ln;

    for (int k0 = 0; k0 < D_MODEL; k0 += 32) {
        v16bf a;
        const uint32_t* arow =
            (const uint32_t*)(ctx + (size_t)mA * D_MODEL + k0);
        #pragma unroll
        for (int p = 0; p < 8; ++p) {
            const int kk = a_koff(p, hf);
            union { uint32_t u; __bf16 hv[2]; } cv;
            cv.u = arow[kk >> 1];
            a[2 * p]     = cv.hv[0];
            a[2 * p + 1] = cv.hv[1];
        }
        #pragma unroll
        for (int nt = 0; nt < 4; ++nt) {
            const int n = n0 + nt * 16 + ln;
            v16bf bm;
            #pragma unroll
            for (int p = 0; p < 8; ++p) {
                const int kk = a_koff(p, hf);
                const float* src = Wo + (size_t)n * D_MODEL + k0 + kk;
                bm[2 * p]     = f2bf(src[0]);
                bm[2 * p + 1] = f2bf(src[1]);
            }
            acc[nt] = wmma_bf16(a, bm, acc[nt]);
        }
    }

    #pragma unroll
    for (int nt = 0; nt < 4; ++nt) {
        const int n = n0 + nt * 16 + ln;
        const float bias = bo[n];
        #pragma unroll
        for (int r = 0; r < 8; ++r) {
            const int row = m0 + r + 8 * hf;
            out[(size_t)row * D_MODEL + n] = acc[nt][r] + bias;
        }
    }
}

// ---------------------------------------------------------------------------
extern "C" void kernel_launch(void* const* d_in, const int* in_sizes, int n_in,
                              void* d_out, int out_size, void* d_ws,
                              size_t ws_size, hipStream_t stream)
{
    (void)in_sizes; (void)n_in; (void)out_size; (void)ws_size;
    const float* Q  = (const float*)d_in[0];
    const float* K  = (const float*)d_in[1];
    const float* V  = (const float*)d_in[2];
    // d_in[3] = causal mask (implicit in kernel)
    const float* Wq = (const float*)d_in[4];
    const float* bq = (const float*)d_in[5];
    const float* Wk = (const float*)d_in[6];
    const float* bk = (const float*)d_in[7];
    const float* Wv = (const float*)d_in[8];
    const float* bv = (const float*)d_in[9];
    const float* Wo = (const float*)d_in[10];
    const float* bo = (const float*)d_in[11];
    float* out = (float*)d_out;

    char* ws = (char*)d_ws;
    const size_t per = (size_t)BATCH * N_HEADS * SEQ * HEAD_DIM * sizeof(__bf16);
    __bf16* qb  = (__bf16*)(ws);
    __bf16* kb  = (__bf16*)(ws + per);
    __bf16* vt  = (__bf16*)(ws + 2 * per);
    __bf16* ctx = (__bf16*)(ws + 3 * per);

    dim3 g1(BS / 16, D_MODEL / 64, 3);
    proj_qkv_kernel<<<g1, 32, 0, stream>>>(Q, K, V, Wq, Wk, Wv, bq, bk, bv,
                                           qb, kb, vt);
    dim3 g2(SEQ / 64, N_HEADS, BATCH);
    attn_kernel<<<g2, 128, 0, stream>>>(qb, kb, vt, ctx);
    dim3 g3(BS / 16, D_MODEL / 64, 1);
    proj_out_kernel<<<g3, 32, 0, stream>>>(ctx, Wo, bo, out);
}
